// ScaledDotProductAttention_3023656976613
// MI455X (gfx1250) — matrix operand order repaired and checked by reading, not compiled, not run on hardware
//
#include <hip/hip_runtime.h>
#include <hip/hip_bf16.h>

// ---------------------------------------------------------------------------
// Scaled-dot-product attention, B=8 H=16 S=512 D=512, fp32 in/out.
// Memory-bound (~8.6 GB effective vs ~69 GFLOP at 23.3 TB/s => ~0.37 ms floor):
// stream q/k/v once (K/V reuse across a head's 8 q-blocks hits the 192MB L2),
// all matmuls via v_wmma_f32_16x16x32_bf16 from LDS, softmax in f32.
// One workgroup = 8 wave32 handles one (b,h) and a 64-row q block.
// global_prefetch_b8 of the next tile overlaps HBM latency with WMMA compute
// (LDS use of ~262KB -> 1 WG/WGP -> only 2 waves/SIMD, so prefetch matters).
// ---------------------------------------------------------------------------

typedef __bf16 bf16;
typedef bf16  v16bf __attribute__((ext_vector_type(16)));
typedef bf16  v4bf  __attribute__((ext_vector_type(4)));
typedef float v8f   __attribute__((ext_vector_type(8)));

constexpr int SEQ = 512;
constexpr int DIM = 512;
constexpr int MB  = 64;    // rows per q block / rows per k,v tile
constexpr int QP  = 520;   // bf16 row stride for q/p and k/v tiles (16B aligned, bank-staggered)
constexpr int SCP = 528;   // f32 row stride for the score block

union FragCast { v16bf v; uint4 u[2]; };

// A fragment (16x32 bf16): lane l -> row (l&15); half (l>>4) selects K-dword
// groups {0..3, 8..11} vs {4..7, 12..15} per the CDNA5 ISA A-matrix table.
__device__ __forceinline__ v16bf load_frag_a(const bf16* base, int row, int kc, int lane) {
  const int h = (lane >> 4) & 1;
  const int r = lane & 15;
  const char* p = (const char*)(base + (size_t)(row + r) * QP) + kc * 64 + h * 16;
  FragCast f;
  f.u[0] = *(const uint4*)p;
  f.u[1] = *(const uint4*)(p + 32);
  return f.v;
}

// B fragment (32x16 bf16): lane l -> column (l&15); half (l>>4) selects K-dword
// groups {0..3, 8..11} vs {4..7, 12..15}, matching the A fragment k order.
// Our B columns are contiguous memory rows (Kᵀ in phase 1, Vᵀ in phase 3).
__device__ __forceinline__ v16bf load_frag_b(const bf16* base, int col0, int kc, int lane) {
  const int h = (lane >> 4) & 1;
  const int r = lane & 15;
  const char* p = (const char*)(base + (size_t)(col0 + r) * QP) + kc * 64 + h * 16;
  FragCast f;
  f.u[0] = *(const uint4*)p;
  f.u[1] = *(const uint4*)(p + 32);
  return f.v;
}

// Load a 64x512 fp32 tile from global, convert to bf16 into LDS (stride QP).
// float4 loads -> fully coalesced 512B per wave per issue.
__device__ __forceinline__ void load_tile_bf16(const float* __restrict__ src,
                                               bf16* __restrict__ dst, int tid) {
  for (int e = tid; e < MB * 128; e += 256) {
    const int r  = e >> 7;      // row 0..63
    const int c4 = e & 127;     // float4 index 0..127
    const float4 f = ((const float4*)(src + (size_t)r * DIM))[c4];
    v4bf hb = { (bf16)f.x, (bf16)f.y, (bf16)f.z, (bf16)f.w };
    *(v4bf*)(dst + (size_t)r * QP + c4 * 4) = hb;
  }
}

// Prefetch a 128KB fp32 tile into cache while WMMA compute runs:
// 256 threads x 2 prefetches at 256B stride -> every cacheline touched.
__device__ __forceinline__ void prefetch_tile(const float* __restrict__ src, int tid) {
  const char* p = (const char*)src + (size_t)tid * 512;
  __builtin_prefetch(p, 0, 1);
  __builtin_prefetch(p + 256, 0, 1);
}

__global__ __launch_bounds__(256) void
attn_b8h16s512d512_kernel(const float* __restrict__ q, const float* __restrict__ k,
                          const float* __restrict__ v, float* __restrict__ out) {
  extern __shared__ char smem[];
  bf16*  qp = (bf16*)smem;                                    // Q tile, reused for P (bf16)
  bf16*  kv = (bf16*)(smem + (size_t)MB * QP * sizeof(bf16)); // K tiles, reused for V tiles
  float* sc = (float*)(smem + (size_t)2 * MB * QP * sizeof(bf16)); // 64x512 scores (f32)

  const int tid  = threadIdx.x;
  const int lane = tid & 31;           // wave32
  const int w    = tid >> 5;           // 8 waves
  const int h    = (lane >> 4) & 1;
  const int r    = lane & 15;

  const int bh    = blockIdx.x >> 3;   // 0..127 = b*16+h
  const int mblk  = blockIdx.x & 7;    // q row block
  const size_t base = (size_t)bh * SEQ * DIM;
  const int m0    = mblk * MB;

  const int mi = w >> 1;               // wave's 16-row subtile (0..3)
  const int nb = (w & 1) * 16;         // wave's first 16-col subtile; second at +32

  load_tile_bf16(q + base + (size_t)m0 * DIM, qp, tid);

  // ---- Phase 1: scores = Q @ Kᵀ (unscaled), full 64x512 block in LDS ----
  for (int kb = 0; kb < 8; ++kb) {
    __syncthreads();                                  // kv free to overwrite
    load_tile_bf16(k + base + (size_t)(kb * MB) * DIM, kv, tid);
    __syncthreads();                                  // kv (and qp on kb==0) ready
    if (kb + 1 < 8)                                   // overlap next K tile's HBM fetch
      prefetch_tile(k + base + (size_t)((kb + 1) * MB) * DIM, tid);
    v8f c0 = {}; v8f c1 = {};
    for (int kc = 0; kc < 16; ++kc) {                 // K-dim 512 = 16 x 32
      v16bf a  = load_frag_a(qp, mi * 16, kc, lane);
      v16bf b0 = load_frag_b(kv, nb,      kc, lane);
      v16bf b1 = load_frag_b(kv, nb + 32, kc, lane);
      c0 = __builtin_amdgcn_wmma_f32_16x16x32_bf16(false, a, false, b0, (short)0, c0, false, false);
      c1 = __builtin_amdgcn_wmma_f32_16x16x32_bf16(false, a, false, b1, (short)0, c1, false, false);
    }
    float* s0 = sc + (size_t)(mi * 16 + h * 8) * SCP + kb * 64 + nb + r;
#pragma unroll
    for (int i = 0; i < 8; ++i) {
      s0[(size_t)i * SCP]      = c0[i];
      s0[(size_t)i * SCP + 32] = c1[i];
    }
  }
  __syncthreads();  // all scores written; qp no longer read as Q

  // ---- Phase 2: row softmax (scale inside), write P as bf16 into qp ----
  {
    prefetch_tile(v + base, tid);              // first V tile during softmax
    const float scale = 0.04419417382415922f;  // 1/sqrt(512)
    const int row = tid >> 2;                  // 4 lanes per row
    const int sub = tid & 3;
    float* srow = sc + (size_t)row * SCP + sub * 128;
    float mx = -3.4e38f;
    for (int j = 0; j < 128; ++j) mx = fmaxf(mx, srow[j]);
    mx = fmaxf(mx, __shfl_xor(mx, 1, 32));
    mx = fmaxf(mx, __shfl_xor(mx, 2, 32));
    float sum = 0.f;
    for (int j = 0; j < 128; ++j) {
      float e = __expf((srow[j] - mx) * scale);  // exp(scale*s - max(scale*s))
      srow[j] = e;
      sum += e;
    }
    sum += __shfl_xor(sum, 1, 32);
    sum += __shfl_xor(sum, 2, 32);
    const float inv = 1.0f / sum;
    bf16* prow = qp + (size_t)row * QP + sub * 128;
    for (int j = 0; j < 128; j += 4) {
      v4bf pb = { (bf16)(srow[j] * inv), (bf16)(srow[j + 1] * inv),
                  (bf16)(srow[j + 2] * inv), (bf16)(srow[j + 3] * inv) };
      *(v4bf*)(prow + j) = pb;
    }
  }

  // ---- Phase 3: out = P @ Vᵀ.  v stored [d][t] => B columns contiguous ----
  float* og = out + base + (size_t)m0 * DIM;
  for (int db = 0; db < 8; ++db) {
    __syncthreads();                                  // kv free; P writes visible
    load_tile_bf16(v + base + (size_t)(db * MB) * DIM, kv, tid);
    __syncthreads();
    if (db + 1 < 8)                                   // overlap next V tile's HBM fetch
      prefetch_tile(v + base + (size_t)((db + 1) * MB) * DIM, tid);
    v8f c0 = {}; v8f c1 = {};
    for (int kc = 0; kc < 16; ++kc) {                 // contraction over t = 512
      v16bf a  = load_frag_a(qp, mi * 16, kc, lane);  // P fragment
      v16bf b0 = load_frag_b(kv, nb,      kc, lane);  // Vᵀ fragment
      v16bf b1 = load_frag_b(kv, nb + 32, kc, lane);
      c0 = __builtin_amdgcn_wmma_f32_16x16x32_bf16(false, a, false, b0, (short)0, c0, false, false);
      c1 = __builtin_amdgcn_wmma_f32_16x16x32_bf16(false, a, false, b1, (short)0, c1, false, false);
    }
    float* o0 = og + (size_t)(mi * 16 + h * 8) * DIM + db * 64 + nb + r;
#pragma unroll
    for (int i = 0; i < 8; ++i) {
      o0[(size_t)i * DIM]      = c0[i];
      o0[(size_t)i * DIM + 32] = c1[i];
    }
  }
}

extern "C" void kernel_launch(void* const* d_in, const int* in_sizes, int n_in,
                              void* d_out, int out_size, void* d_ws, size_t ws_size,
                              hipStream_t stream) {
  (void)in_sizes; (void)n_in; (void)out_size; (void)d_ws; (void)ws_size;
  const float* q = (const float*)d_in[0];
  const float* k = (const float*)d_in[1];
  const float* v = (const float*)d_in[2];
  float* out = (float*)d_out;

  const size_t shmem = (size_t)2 * MB * QP * sizeof(bf16)   // Q/P + K/V tiles
                     + (size_t)MB * SCP * sizeof(float);    // score block  (~262 KB of 320 KB)
  // Allow > default dynamic LDS (WGP has 320 KB). Idempotent; no static guards
  // (harness requires deterministic, stateless kernel_launch).
  (void)hipFuncSetAttribute((const void*)attn_b8h16s512d512_kernel,
                            hipFuncAttributeMaxDynamicSharedMemorySize, (int)shmem);

  dim3 grid(8 * 16 * (SEQ / MB));  // 1024 workgroups: (b,h) x q-row-block
  dim3 block(256);                 // 8 wave32
  attn_b8h16s512d512_kernel<<<grid, block, shmem, stream>>>(q, k, v, out);
}
